// RecurrentFFN_33535104647391
// MI455X (gfx1250) — compile-verified, matrix operand order checked by reading, not run on hardware
//
#include <hip/hip_runtime.h>
#include <hip/hip_bf16.h>

// Problem dims (from reference)
#define B_  8
#define T_  64
#define D_  1024
#define H_  4096
#define K_  (D_ + H_)      // 5120 = concat(x, h) feature dim
#define EPSF 1e-3f

// Persistent-kernel shape: 64 blocks x 256 threads = 512 wave32 waves,
// exactly matching phase A's 512 output tiles (z: 256, r: 256).
#define NB_ 64
#define NT_ 256

typedef __bf16 bf16_t;
typedef __attribute__((ext_vector_type(16))) __bf16 v16bf;
typedef __attribute__((ext_vector_type(8)))  float  v8f;

union FragU { uint4 u[2]; v16bf v; };

// Load one wave32 WMMA 16x32 bf16 operand fragment for this lane.
// Per CDNA5 ISA layout (16-bit A 16x32): lanes 0-15 hold K=0..7 (vgpr0-3) and
// K=16..23 (vgpr4-7); lanes 16-31 hold K=8..15 and K=24..31. That is two
// contiguous 8-element (16-byte) runs per lane -> two b128 loads.
__device__ __forceinline__ v16bf load_frag16(const bf16_t* __restrict__ row,
                                             int k0, int half) {
  FragU f;
  f.u[0] = *reinterpret_cast<const uint4*>(row + k0 + half * 8);
  f.u[1] = *reinterpret_cast<const uint4*>(row + k0 + 16 + half * 8);
  return f.v;
}

__device__ __forceinline__ float sigmoidf_(float v) {
  return 1.0f / (1.0f + __expf(-v));
}

// Grid-wide barrier: monotonic counter, release fence before arrive,
// acquire fence after the spin. Counter is hipMemsetAsync'd to 0 each launch
// so the kernel is stateless across calls. 64 blocks are co-resident by
// construction (one tiny block per WGP at most), so the spin cannot deadlock.
__device__ __forceinline__ void grid_sync(unsigned* counter, unsigned target) {
  __syncthreads();
  if (threadIdx.x == 0) {
    __threadfence();                      // release: publish phase writes
    atomicAdd(counter, 1u);
    while (__hip_atomic_load(counter, __ATOMIC_RELAXED,
                             __HIP_MEMORY_SCOPE_AGENT) < target) {
      __builtin_amdgcn_s_sleep(1);
    }
    __threadfence();                      // acquire: invalidate stale lines
  }
  __syncthreads();
}

// ---------------------------------------------------------------------------
// One-time per launch: W[K][N] fp32 (row-major)  ->  Wt[N][K] bf16.
// Pre-transposing makes the WMMA B-operand (a column of W) a contiguous row,
// so B fragments are two b128 loads, and bf16 weights (168 MB total) become
// L2-resident (192 MB L2) across all 64 timesteps.
// ---------------------------------------------------------------------------
__global__ void k_transpose_bf16(const float* __restrict__ src,
                                 bf16_t* __restrict__ dst, int K, int N) {
  __shared__ float tile[32][33];
  const int gx = blockIdx.x * 32;  // N direction
  const int gy = blockIdx.y * 32;  // K direction
#pragma unroll
  for (int i = 0; i < 4; ++i) {
    int k = gy + threadIdx.y + i * 8;
    tile[threadIdx.y + i * 8][threadIdx.x] =
        src[(size_t)k * N + (gx + threadIdx.x)];
  }
  __syncthreads();
#pragma unroll
  for (int i = 0; i < 4; ++i) {
    int n = gx + threadIdx.y + i * 8;
    dst[(size_t)n * K + (gy + threadIdx.x)] =
        (bf16_t)tile[threadIdx.x][threadIdx.y + i * 8];
  }
}

// ---------------------------------------------------------------------------
// Persistent cooperative kernel: the entire 64-step recurrence in ONE launch.
// Every phase keeps ALL 64 blocks (all WGPs / full L2 bandwidth) busy:
//   A: 512 tiles over 512 waves
//   B: 256 tiles x 2-way K-split over 512 waves (LDS partial reduce)
//   D: 64 tiles (1/block) x 8-way K-split (LDS partial reduce)
// ---------------------------------------------------------------------------
__global__ void __launch_bounds__(NT_, 1)
k_recurrent(const float* __restrict__ x, const float* __restrict__ h0,
            const bf16_t* __restrict__ Wzt, const float* __restrict__ bz,
            const bf16_t* __restrict__ Wrt, const float* __restrict__ br,
            const bf16_t* __restrict__ Wgt, const float* __restrict__ bg,
            const bf16_t* __restrict__ Wut, const float* __restrict__ bu,
            const bf16_t* __restrict__ Wdt, const float* __restrict__ bd,
            const float* __restrict__ gh, const float* __restrict__ bh,
            const float* __restrict__ go, const float* __restrict__ bo,
            bf16_t* __restrict__ comb, bf16_t* __restrict__ cand,
            float* __restrict__ zbuf, float* __restrict__ hbuf,
            float* __restrict__ hpre, float* __restrict__ opre,
            float* __restrict__ out, unsigned* counter) {
  __shared__ float s_sum[NT_];
  __shared__ float s_sq[NT_];
  __shared__ float s_red[2048];                    // 8 KB partial-acc buffer
  const int tid  = threadIdx.x;
  const int lane = tid & 31;
  const int wv   = tid >> 5;
  const int gw   = blockIdx.x * (NT_ >> 5) + wv;   // global wave id, 0..511
  const int half = lane >> 4;
  const int lr   = lane & 15;
  const int tg   = blockIdx.x * NT_ + tid;         // global thread id
  unsigned epoch = 0;

  // ---- init: h0 -> fp32 state + bf16 h-region of comb ----
  for (int i = tg; i < B_ * H_; i += NB_ * NT_) {
    int b = i >> 12;           // / H_
    int j = i & (H_ - 1);
    float v = h0[i];
    hbuf[i] = v;
    comb[b * K_ + D_ + j] = (bf16_t)v;
  }

  for (int t = 0; t < T_; ++t) {
    // ---- prep: x[:, t, :] -> bf16 into x-region of comb and cand ----
    if (tg < B_ * D_) {
      int b = tg >> 10;        // / D_
      int d = tg & (D_ - 1);
      bf16_t v = (bf16_t)x[((size_t)b * T_ + t) * D_ + d];
      comb[b * K_ + d] = v;
      cand[b * K_ + d] = v;
    }
    grid_sync(counter, ++epoch * NB_);

    // ---- phase A: z = sig(comb@Wz+bz) [gw<256], r -> r*h into cand [gw>=256]
    {
      const bool is_r = (gw >= 256);
      const int tile_n = (gw & 255) * 16;
      const bf16_t* Wt   = is_r ? Wrt : Wzt;
      const float*  bias = is_r ? br  : bz;
      const bf16_t* arow = comb + (size_t)lr * K_;
      const bf16_t* brow = Wt + (size_t)(tile_n + lr) * K_;
      v8f acc = {};
      for (int k0 = 0; k0 < K_; k0 += 32) {
        v16bf a = load_frag16(arow, k0, half);
        v16bf b = load_frag16(brow, k0, half);
        acc = __builtin_amdgcn_wmma_f32_16x16x32_bf16(false, a, false, b,
                                                      (short)0, acc,
                                                      false, false);
      }
      if (half == 0) {           // rows 0..7 = real batch rows
        const int col = tile_n + lr;
        const float bv = bias[col];
#pragma unroll
        for (int j = 0; j < 8; ++j) {
          float g = sigmoidf_(acc[j] + bv);
          if (is_r) {
            cand[j * K_ + D_ + col] = (bf16_t)(g * hbuf[j * H_ + col]);
          } else {
            zbuf[j * H_ + col] = g;
          }
        }
      }
    }
    grid_sync(counter, ++epoch * NB_);

    // ---- phase B: gate/up fused (shared A, dual accumulators), 2-way
    //      K-split so all 64 blocks stream weights; then
    //      hpre = (1-z)*h + z*(silu(gate)*up)
    {
      const int p  = wv & 3;                       // tile slot in block
      const int kh = wv >> 2;                      // K-half (0 or 1)
      const int tile_n = (blockIdx.x * 4 + p) * 16;
      const bf16_t* arow = cand + (size_t)lr * K_;
      const bf16_t* grow = Wgt + (size_t)(tile_n + lr) * K_;
      const bf16_t* urow = Wut + (size_t)(tile_n + lr) * K_;
      v8f accG = {};
      v8f accU = {};
      const int kbeg = kh * (K_ / 2);
      const int kend = kbeg + (K_ / 2);
      for (int k0 = kbeg; k0 < kend; k0 += 32) {
        v16bf a   = load_frag16(arow, k0, half);
        v16bf bg_ = load_frag16(grow, k0, half);
        v16bf bu_ = load_frag16(urow, k0, half);
        accG = __builtin_amdgcn_wmma_f32_16x16x32_bf16(false, a, false, bg_,
                                                       (short)0, accG,
                                                       false, false);
        accU = __builtin_amdgcn_wmma_f32_16x16x32_bf16(false, a, false, bu_,
                                                       (short)0, accU,
                                                       false, false);
      }
      if (kh == 1) {
        *(v8f*)&s_red[((p * 2 + 0) * 32 + lane) * 8] = accG;
        *(v8f*)&s_red[((p * 2 + 1) * 32 + lane) * 8] = accU;
      }
      __syncthreads();
      if (kh == 0) {
        accG += *(const v8f*)&s_red[((p * 2 + 0) * 32 + lane) * 8];
        accU += *(const v8f*)&s_red[((p * 2 + 1) * 32 + lane) * 8];
        if (half == 0) {
          const int col = tile_n + lr;
          const float bgv = bg[col];
          const float buv = bu[col];
#pragma unroll
          for (int j = 0; j < 8; ++j) {
            float g  = accG[j] + bgv;
            float u  = accU[j] + buv;
            float sw = (g * sigmoidf_(g)) * u;     // silu(gate) * up
            float z  = zbuf[j * H_ + col];
            float hv = hbuf[j * H_ + col];
            hpre[j * H_ + col] = (1.0f - z) * hv + z * sw;
          }
        }
      }
    }
    grid_sync(counter, ++epoch * NB_);

    // ---- LayerNorm over H per batch row (blocks 0..7), write fp32 state
    //      and bf16 into comb's h-region
    if (blockIdx.x < B_) {
      const int b = blockIdx.x;
      const float* row = hpre + (size_t)b * H_;
      float sum = 0.0f, sq = 0.0f;
      for (int j = tid; j < H_; j += NT_) {
        float v = row[j];
        sum += v;
        sq += v * v;
      }
      s_sum[tid] = sum;
      s_sq[tid] = sq;
      __syncthreads();
      for (int off = NT_ / 2; off > 0; off >>= 1) {
        if (tid < off) {
          s_sum[tid] += s_sum[tid + off];
          s_sq[tid]  += s_sq[tid + off];
        }
        __syncthreads();
      }
      const float mean = s_sum[0] * (1.0f / H_);
      const float var  = s_sq[0] * (1.0f / H_) - mean * mean;
      const float inv  = rsqrtf(var + EPSF);
      for (int j = tid; j < H_; j += NT_) {
        float v = (row[j] - mean) * inv * gh[j] + bh[j];
        hbuf[b * H_ + j] = v;
        comb[b * K_ + D_ + j] = (bf16_t)v;
      }
    }
    grid_sync(counter, ++epoch * NB_);

    // ---- phase D: opre = h @ Wd + bd. One tile per block, 8-way K-split
    //      (512 K per wave), LDS reduction by wave 0.
    {
      const int tile_n = blockIdx.x * 16;          // 64 tiles over 64 blocks
      const bf16_t* arow = comb + (size_t)lr * K_ + D_;   // h row, 4096 bf16
      const bf16_t* brow = Wdt + (size_t)(tile_n + lr) * H_;
      v8f acc = {};
      const int kbeg = wv * (H_ / 8);
      const int kend = kbeg + (H_ / 8);
      for (int k0 = kbeg; k0 < kend; k0 += 32) {
        v16bf a = load_frag16(arow, k0, half);
        v16bf b = load_frag16(brow, k0, half);
        acc = __builtin_amdgcn_wmma_f32_16x16x32_bf16(false, a, false, b,
                                                      (short)0, acc,
                                                      false, false);
      }
      if (wv != 0) {
        *(v8f*)&s_red[((wv - 1) * 32 + lane) * 8] = acc;
      }
      __syncthreads();
      if (wv == 0) {
#pragma unroll
        for (int w = 0; w < 7; ++w) {
          acc += *(const v8f*)&s_red[(w * 32 + lane) * 8];
        }
        if (half == 0) {
          const int col = tile_n + lr;
          const float bv = bd[col];
#pragma unroll
          for (int j = 0; j < 8; ++j) {
            opre[j * D_ + col] = acc[j] + bv;
          }
        }
      }
    }
    grid_sync(counter, ++epoch * NB_);

    // ---- LayerNorm over D per batch row (blocks 0..7), write out[b, t, :].
    //      Next iteration's first barrier orders this against step t+1.
    if (blockIdx.x < B_) {
      const int b = blockIdx.x;
      const float* row = opre + (size_t)b * D_;
      float sum = 0.0f, sq = 0.0f;
      for (int d = tid; d < D_; d += NT_) {
        float v = row[d];
        sum += v;
        sq += v * v;
      }
      s_sum[tid] = sum;
      s_sq[tid] = sq;
      __syncthreads();
      for (int off = NT_ / 2; off > 0; off >>= 1) {
        if (tid < off) {
          s_sum[tid] += s_sum[tid + off];
          s_sq[tid]  += s_sq[tid + off];
        }
        __syncthreads();
      }
      const float mean = s_sum[0] * (1.0f / D_);
      const float var  = s_sq[0] * (1.0f / D_) - mean * mean;
      const float inv  = rsqrtf(var + EPSF);
      float* orow = out + ((size_t)b * T_ + t) * D_;
      for (int d = tid; d < D_; d += NT_) {
        orow[d] = (row[d] - mean) * inv * go[d] + bo[d];
      }
    }
  }
}

// ---------------------------------------------------------------------------
extern "C" void kernel_launch(void* const* d_in, const int* in_sizes, int n_in,
                              void* d_out, int out_size, void* d_ws,
                              size_t ws_size, hipStream_t stream) {
  (void)in_sizes; (void)n_in; (void)out_size; (void)ws_size;
  const float* x  = (const float*)d_in[0];
  const float* h0 = (const float*)d_in[1];
  const float* Wz = (const float*)d_in[2];  const float* bz = (const float*)d_in[3];
  const float* Wr = (const float*)d_in[4];  const float* br = (const float*)d_in[5];
  const float* Wg = (const float*)d_in[6];  const float* bg = (const float*)d_in[7];
  const float* Wu = (const float*)d_in[8];  const float* bu = (const float*)d_in[9];
  const float* Wd = (const float*)d_in[10]; const float* bd = (const float*)d_in[11];
  const float* gh = (const float*)d_in[12]; const float* bh = (const float*)d_in[13];
  const float* go = (const float*)d_in[14]; const float* bo = (const float*)d_in[15];
  float* out = (float*)d_out;

  // Workspace carve (all 256-byte aligned)
  char* ws = (char*)d_ws;
  auto carve = [&](size_t bytes) -> char* {
    char* p = ws;
    ws += (bytes + 255) & ~(size_t)255;
    return p;
  };
  unsigned* counter = (unsigned*)carve(256);                       // barrier
  bf16_t* Wzt = (bf16_t*)carve((size_t)H_ * K_ * sizeof(bf16_t));  // [4096][5120]
  bf16_t* Wrt = (bf16_t*)carve((size_t)H_ * K_ * sizeof(bf16_t));
  bf16_t* Wgt = (bf16_t*)carve((size_t)H_ * K_ * sizeof(bf16_t));
  bf16_t* Wut = (bf16_t*)carve((size_t)H_ * K_ * sizeof(bf16_t));
  bf16_t* Wdt = (bf16_t*)carve((size_t)D_ * H_ * sizeof(bf16_t));  // [1024][4096]
  bf16_t* comb = (bf16_t*)carve((size_t)16 * K_ * sizeof(bf16_t)); // rows 8-15 don't-care
  bf16_t* cand = (bf16_t*)carve((size_t)16 * K_ * sizeof(bf16_t));
  float* zbuf = (float*)carve((size_t)B_ * H_ * sizeof(float));
  float* hbuf = (float*)carve((size_t)B_ * H_ * sizeof(float));
  float* hpre = (float*)carve((size_t)B_ * H_ * sizeof(float));
  float* opre = (float*)carve((size_t)B_ * D_ * sizeof(float));

  // Stateless barrier counter (graph-capture-legal async memset)
  hipMemsetAsync(counter, 0, 256, stream);

  // One-time bf16 transpose of all weights (~168 MB -> L2-resident thereafter)
  dim3 tb(32, 8);
  k_transpose_bf16<<<dim3(H_ / 32, K_ / 32), tb, 0, stream>>>(Wz, Wzt, K_, H_);
  k_transpose_bf16<<<dim3(H_ / 32, K_ / 32), tb, 0, stream>>>(Wr, Wrt, K_, H_);
  k_transpose_bf16<<<dim3(H_ / 32, K_ / 32), tb, 0, stream>>>(Wg, Wgt, K_, H_);
  k_transpose_bf16<<<dim3(H_ / 32, K_ / 32), tb, 0, stream>>>(Wu, Wut, K_, H_);
  k_transpose_bf16<<<dim3(D_ / 32, H_ / 32), tb, 0, stream>>>(Wd, Wdt, H_, D_);

  // Entire 64-step recurrence: one persistent cooperative kernel.
  k_recurrent<<<NB_, NT_, 0, stream>>>(x, h0, Wzt, bz, Wrt, br, Wgt, bg,
                                       Wut, bu, Wdt, bd, gh, bh, go, bo,
                                       comb, cand, zbuf, hbuf, hpre, opre,
                                       out, counter);
}
